// EnhancedBrain_52415780880484
// MI455X (gfx1250) — matrix operand order repaired and checked by reading, not compiled, not run on hardware
//
#include <hip/hip_runtime.h>
#include <hip/hip_bf16.h>

// ---------------- problem constants (match reference) ----------------
#define D_MODEL 1024
#define D_FF    4096
#define N_ZONES 8
#define TOP_K   3
#define ROUTER_H 256
#define BATCH   16
#define SEQ     2048

// ---------------- vector types for WMMA ----------------
typedef __attribute__((ext_vector_type(16))) __bf16          v16bf;
typedef __attribute__((ext_vector_type(16))) unsigned short  v16u;
typedef __attribute__((ext_vector_type(8)))  unsigned short  u16x8;
typedef __attribute__((ext_vector_type(4)))  unsigned short  u16x4;
typedef __attribute__((ext_vector_type(2)))  unsigned int    u32x2;
typedef __attribute__((ext_vector_type(8)))  float           v8f;

union BfCast { v16u u; v16bf b; };

// fp32 pair -> packed bf16 (RNE). Uses v_cvt_pk_bf16_f32 when the toolchain
// declares it for gfx1250, else 2-op bit-twiddled RNE per element.
__device__ __forceinline__ unsigned pack2bf(float a, float b) {
#if __has_builtin(__builtin_amdgcn_cvt_pk_bf16_f32)
    typedef __attribute__((ext_vector_type(2))) __bf16 v2bf;
    v2bf r = __builtin_amdgcn_cvt_pk_bf16_f32(a, b);
    return __builtin_bit_cast(unsigned, r);
#else
    unsigned ua = __builtin_bit_cast(unsigned, a);
    ua += 0x7FFFu + ((ua >> 16) & 1u);
    unsigned ub = __builtin_bit_cast(unsigned, b);
    ub += 0x7FFFu + ((ub >> 16) & 1u);
    return (ua >> 16) | (ub & 0xFFFF0000u);
#endif
}

// branchless tanh-approx gelu: tanh(z) = 1 - 2*rcp(exp(2z)+1)
// (v_exp_f32 + v_rcp_f32; saturates correctly at +/-inf, no EXEC divergence)
__device__ __forceinline__ float gelu_tanh(float v) {
    const float k0 = 0.7978845608028654f;  // sqrt(2/pi)
    const float k1 = 0.044715f;
    float z  = k0 * (v + k1 * v * v * v);
    float ez = __expf(2.0f * z);
    float th = 1.0f - 2.0f * __builtin_amdgcn_rcpf(ez + 1.0f);
    return 0.5f * v * (1.0f + th);
}

// A fragment (16x32 bf16, row-major source in LDS, row stride in ushorts).
// ISA 7.12.2: lanes 0-15 -> K {0..7,16..23}, lanes 16-31 -> K {8..15,24..31}; M = lane&15.
__device__ __forceinline__ v16bf load_a_frag(const unsigned short* base, int stride,
                                             int row0, int kb, int lane) {
    int M = lane & 15, hlf = lane >> 4;
    const unsigned short* rp = base + (size_t)(row0 + M) * stride + kb;
    u16x8 lo = *(const u16x8*)(rp + hlf * 8);
    u16x8 hi = *(const u16x8*)(rp + 16 + hlf * 8);
    BfCast c;
    c.u = __builtin_shufflevector(lo, hi, 0,1,2,3,4,5,6,7,8,9,10,11,12,13,14,15);
    return c.b;
}

// B fragment (32x16 bf16) from an LDS tile stored TRANSPOSED as [ncols][64] (K contiguous).
// ISA: lanes 0-15 hold K {0..7,16..23}, lanes 16-31 hold K {8..15,24..31}; N = lane&15.
__device__ __forceinline__ v16bf load_b_frag_t(const unsigned short* Wt,
                                               int n0, int kb, int lane) {
    int N = lane & 15, hlf = lane >> 4;
    const unsigned short* cp = Wt + (size_t)(n0 + N) * 64 + kb + hlf * 8;
    u16x8 lo = *(const u16x8*)(cp);
    u16x8 hi = *(const u16x8*)(cp + 16);
    BfCast c;
    c.u = __builtin_shufflevector(lo, hi, 0,1,2,3,4,5,6,7,8,9,10,11,12,13,14,15);
    return c.b;
}

__device__ __forceinline__ v8f wmma_bf16(v16bf a, v16bf b, v8f c) {
    return __builtin_amdgcn_wmma_f32_16x16x32_bf16(false, a, false, b,
                                                   (short)0, c, false, false);
}

// ---------------- kernel 1: mean pool over sequence ----------------
__global__ __launch_bounds__(256) void eb_pool_kernel(const float* __restrict__ x,
                                                      float* __restrict__ pooled) {
    int b = blockIdx.x;
    const float* xb = x + (size_t)b * SEQ * D_MODEL;
    for (int c = threadIdx.x; c < D_MODEL; c += 256) {
        float s = 0.0f;
        for (int r = 0; r < SEQ; ++r) s += xb[(size_t)r * D_MODEL + c];
        pooled[b * D_MODEL + c] = s * (1.0f / (float)SEQ);
    }
}

// ---------------- kernel 2: router MLP + softmax + top-3 ----------------
__global__ __launch_bounds__(256) void eb_router_kernel(const float* __restrict__ pooled,
                                                        const float* __restrict__ W1,
                                                        const float* __restrict__ b1,
                                                        const float* __restrict__ W2,
                                                        const float* __restrict__ b2,
                                                        float* __restrict__ selw,
                                                        int*   __restrict__ seli) {
    int b = blockIdx.x, t = threadIdx.x;
    __shared__ float hsh[ROUTER_H];
    __shared__ float lsh[N_ZONES];
    const float* p = pooled + b * D_MODEL;
    float s = 0.0f;
    for (int i = 0; i < D_MODEL; ++i) s += p[i] * W1[i * ROUTER_H + t];
    hsh[t] = tanhf(s + b1[t]);
    __syncthreads();
    if (t < N_ZONES) {
        float l = b2[t];
        for (int i = 0; i < ROUTER_H; ++i) l += hsh[i] * W2[i * N_ZONES + t];
        lsh[t] = l;
    }
    __syncthreads();
    if (t == 0) {
        float mx = lsh[0];
        for (int e = 1; e < N_ZONES; ++e) mx = fmaxf(mx, lsh[e]);
        float v[N_ZONES];
        for (int e = 0; e < N_ZONES; ++e) v[e] = __expf(lsh[e] - mx);
        // top-3 (softmax denominator cancels in the renormalization)
        bool used[N_ZONES] = {false};
        float tw[TOP_K]; int ti[TOP_K];
        for (int k = 0; k < TOP_K; ++k) {
            float bv = -1.0f; int bi = 0;
            for (int e = 0; e < N_ZONES; ++e)
                if (!used[e] && v[e] > bv) { bv = v[e]; bi = e; }
            used[bi] = true; tw[k] = bv; ti[k] = bi;
        }
        float tsum = tw[0] + tw[1] + tw[2];
        for (int k = 0; k < TOP_K; ++k) {
            selw[b * TOP_K + k] = tw[k] / tsum;
            seli[b * TOP_K + k] = ti[k];
        }
    }
}

// ---------------- kernel 3: fused top-3 expert MLP, WMMA bf16 ----------------
// grid = (SEQ/64, BATCH), block = 256 threads = 8 wave32s.
// Per block: 64-token tile; out[64][1024] f32 lives in registers
// (8 waves x 32 tiles of 16x16 = 256 accumulator VGPRs per wave).
#define MT 64          // tokens per block
#define FC 64          // D_FF chunk
#define KK 64          // D_MODEL K chunk for GEMM1
#define NC 256         // D_MODEL N chunk for GEMM2

__global__ __launch_bounds__(256) void eb_moe_kernel(const float* __restrict__ x,
                                                     const float* __restrict__ Wa,
                                                     const float* __restrict__ Wb,
                                                     const float* __restrict__ selw,
                                                     const int*   __restrict__ seli,
                                                     float* __restrict__ out) {
    __shared__ __align__(16) unsigned short Xs[MT * KK];       // 8 KB  (row-major [64][64])
    __shared__ __align__(16) unsigned short WaT[FC * KK];      // 8 KB  (transposed [fc][k])
    __shared__ __align__(16) unsigned short Hs[MT * FC];       // 8 KB  (row-major [64][64])
    __shared__ __align__(16) unsigned short WbT[NC * FC];      // 32 KB (transposed [n][k=fc])

    const int b   = blockIdx.y;
    const int s0  = blockIdx.x * MT;
    const int tid = threadIdx.x;
    const int lane = tid & 31;
    const int wv   = tid >> 5;
    const int hlf  = lane >> 4;
    const int Mn   = lane & 15;      // doubles as M (A/C rows) and N (B/C cols)

    // GEMM1 roles: H is 64x64 -> 4x4 tiles, two per wave
    const int hr = wv & 3;           // H row-tile (0..3)
    const int hc = (wv >> 2) * 2;    // H col-tile pair base (0 or 2)
    // GEMM2 roles: within a 256-col N-chunk: 4 row-tiles x 16 col-tiles, 8 per wave
    const int r2 = wv & 3;           // out row-tile (0..3)
    const int cg = (wv >> 2) * 8;    // col-tile octet base (0 or 8)

    const float* xblk = x + ((size_t)b * SEQ + s0) * D_MODEL;

    v8f acc[32];                     // [nc][jj] : 4 x 8
    v8f vzero = {};
#pragma unroll
    for (int i = 0; i < 32; ++i) acc[i] = vzero;

    for (int k = 0; k < TOP_K; ++k) {
        const int   e   = seli[b * TOP_K + k];
        const float wgt = selw[b * TOP_K + k];
        const float* WaE = Wa + (size_t)e * D_MODEL * D_FF;
        const float* WbE = Wb + (size_t)e * D_FF * D_MODEL;

        for (int fc = 0; fc < D_FF; fc += FC) {
            __builtin_prefetch(&WbE[(size_t)fc * D_MODEL], 0, 1);  // global_prefetch_b8
            v8f hacc[2] = {vzero, vzero};

            // ---- GEMM1: H[64,64] = X[64,1024] @ Wa[:, fc:fc+64] ----
            for (int kk = 0; kk < D_MODEL; kk += KK) {
                // stage X chunk [64][64] f32 -> bf16 (float4 loads, packed stores)
#pragma unroll
                for (int j = 0; j < 4; ++j) {
                    int idx4 = tid + 256 * j;              // 1024 float4 groups
                    int r = idx4 >> 4, c0 = (idx4 * 4) & 63;
                    float4 vv = *(const float4*)&xblk[(size_t)r * D_MODEL + kk + c0];
                    u32x2 pk = { pack2bf(vv.x, vv.y), pack2bf(vv.z, vv.w) };
                    *(u32x2*)&Xs[r * KK + c0] = pk;
                }
                // stage Wa chunk [64][64] transposed into WaT[fc-local][k]
#pragma unroll
                for (int j = 0; j < 4; ++j) {
                    int idx4 = tid + 256 * j;              // 1024 float4 groups
                    int r = idx4 >> 4, c0 = (idx4 * 4) & 63;
                    float4 vv = *(const float4*)&WaE[(size_t)(kk + r) * D_FF + fc + c0];
                    unsigned p01 = pack2bf(vv.x, vv.y);
                    unsigned p23 = pack2bf(vv.z, vv.w);
                    WaT[(c0 + 0) * KK + r] = (unsigned short)p01;
                    WaT[(c0 + 1) * KK + r] = (unsigned short)(p01 >> 16);
                    WaT[(c0 + 2) * KK + r] = (unsigned short)p23;
                    WaT[(c0 + 3) * KK + r] = (unsigned short)(p23 >> 16);
                }
                __syncthreads();
#pragma unroll
                for (int ks = 0; ks < KK; ks += 32) {
                    v16bf af = load_a_frag(Xs, KK, hr * 16, ks, lane);
                    v16bf bf0 = load_b_frag_t(WaT, (hc + 0) * 16, ks, lane);
                    v16bf bf1 = load_b_frag_t(WaT, (hc + 1) * 16, ks, lane);
                    hacc[0] = wmma_bf16(af, bf0, hacc[0]);
                    hacc[1] = wmma_bf16(af, bf1, hacc[1]);
                }
                __syncthreads();
            }

            // ---- gelu * weight -> Hs (bf16, packed conversion) ----
#pragma unroll
            for (int t = 0; t < 2; ++t) {
#pragma unroll
                for (int v = 0; v < 8; v += 2) {
                    float g0 = gelu_tanh(hacc[t][v + 0]) * wgt;
                    float g1 = gelu_tanh(hacc[t][v + 1]) * wgt;
                    unsigned p = pack2bf(g0, g1);
                    int m0 = v + 0 + 8 * hlf;              // C-layout rows
                    int m1 = v + 1 + 8 * hlf;
                    int colc = (hc + t) * 16 + Mn;
                    Hs[(hr * 16 + m0) * FC + colc] = (unsigned short)p;
                    Hs[(hr * 16 + m1) * FC + colc] = (unsigned short)(p >> 16);
                }
            }
            __syncthreads();

            // ---- GEMM2: out[64,1024] += Hs[64,64] @ Wb[fc:fc+64, :] ----
            for (int nc = 0; nc < 4; ++nc) {
                // stage Wb chunk [64][256] transposed into WbT[n][k]
#pragma unroll
                for (int j = 0; j < 16; ++j) {
                    int idx4 = tid + 256 * j;              // 4096 float4 groups
                    int r = idx4 >> 6, c0 = (idx4 * 4) & 255;
                    float4 vv = *(const float4*)&WbE[(size_t)(fc + r) * D_MODEL + nc * NC + c0];
                    unsigned p01 = pack2bf(vv.x, vv.y);
                    unsigned p23 = pack2bf(vv.z, vv.w);
                    WbT[(c0 + 0) * FC + r] = (unsigned short)p01;
                    WbT[(c0 + 1) * FC + r] = (unsigned short)(p01 >> 16);
                    WbT[(c0 + 2) * FC + r] = (unsigned short)p23;
                    WbT[(c0 + 3) * FC + r] = (unsigned short)(p23 >> 16);
                }
                __syncthreads();
#pragma unroll
                for (int ks = 0; ks < FC; ks += 32) {
                    v16bf af = load_a_frag(Hs, FC, r2 * 16, ks, lane);
                    // batch-load all 8 B fragments first, then run the WMMAs
                    // back-to-back so one dscnt wait covers the whole group
                    v16bf bfr[8];
#pragma unroll
                    for (int jj = 0; jj < 8; ++jj)
                        bfr[jj] = load_b_frag_t(WbT, (cg + jj) * 16, ks, lane);
#pragma unroll
                    for (int jj = 0; jj < 8; ++jj)
                        acc[nc * 8 + jj] = wmma_bf16(af, bfr[jj], acc[nc * 8 + jj]);
                }
                __syncthreads();
            }
        }
    }

    // ---- epilogue: out = x + weighted zone sum (non-temporal; out is
    // write-once and this is the last read of these x rows -> keep L2 for
    // the Wa/Wb working set other blocks are reusing) ----
#pragma unroll
    for (int nc = 0; nc < 4; ++nc) {
#pragma unroll
        for (int jj = 0; jj < 8; ++jj) {
            int col = nc * NC + (cg + jj) * 16 + Mn;
#pragma unroll
            for (int v = 0; v < 8; ++v) {
                int row = r2 * 16 + v + 8 * hlf;
                size_t off = ((size_t)b * SEQ + s0 + row) * D_MODEL + col;
                float xv = __builtin_nontemporal_load(&x[off]);
                __builtin_nontemporal_store(xv + acc[nc * 8 + jj][v], &out[off]);
            }
        }
    }
}

// ---------------- launcher ----------------
extern "C" void kernel_launch(void* const* d_in, const int* in_sizes, int n_in,
                              void* d_out, int out_size, void* d_ws, size_t ws_size,
                              hipStream_t stream) {
    const float* x  = (const float*)d_in[0];
    const float* W1 = (const float*)d_in[1];
    const float* b1 = (const float*)d_in[2];
    const float* W2 = (const float*)d_in[3];
    const float* b2 = (const float*)d_in[4];
    const float* Wa = (const float*)d_in[5];
    const float* Wb = (const float*)d_in[6];
    float* out = (float*)d_out;

    float* wsf    = (float*)d_ws;
    float* pooled = wsf;                              // 16*1024 floats
    float* selw   = wsf + BATCH * D_MODEL;            // 16*3 floats (padded to 64)
    int*   seli   = (int*)(wsf + BATCH * D_MODEL + 64);

    eb_pool_kernel  <<<BATCH, 256, 0, stream>>>(x, pooled);
    eb_router_kernel<<<BATCH, 256, 0, stream>>>(pooled, W1, b1, W2, b2, selw, seli);
    eb_moe_kernel   <<<dim3(SEQ / MT, BATCH), 256, 0, stream>>>(x, Wa, Wb, selw, seli, out);
}
